// Attention_51110110823042
// MI455X (gfx1250) — matrix operand order repaired and checked by reading, not compiled, not run on hardware
//
#include <hip/hip_runtime.h>

// ---- problem constants (match reference) ----
#define B_SZ 2
#define SEQ  2048
#define EMB  1024
#define NH   16
#define HD   64    // head dim; scale = 1/8

typedef __attribute__((ext_vector_type(16))) __bf16 bf16x16;
typedef __attribute__((ext_vector_type(8)))  __bf16 bf16x8;
typedef __attribute__((ext_vector_type(8)))  float  f32x8;

// ---- CDNA5 async global->LDS path (probe-confirmed builtin; guarded) ----
#if defined(__HIP_DEVICE_COMPILE__) && __has_builtin(__builtin_amdgcn_global_load_async_to_lds_b128)
#define HAVE_ASYNC 1
#else
#define HAVE_ASYNC 0
#endif

typedef int v4i __attribute__((vector_size(16)));
typedef __attribute__((address_space(1))) v4i GV4;   // global 16B chunk
typedef __attribute__((address_space(3))) v4i LV4;   // LDS    16B chunk

__device__ __forceinline__ void cp16_g2l(void* lds_dst, const void* gsrc) {
#if HAVE_ASYNC
  __builtin_amdgcn_global_load_async_to_lds_b128((GV4*)gsrc, (LV4*)lds_dst, 0, 0);
#else
  *(bf16x8*)lds_dst = *(const bf16x8*)gsrc;
#endif
}

__device__ __forceinline__ void cp_wait_all() {
#if HAVE_ASYNC
#if __has_builtin(__builtin_amdgcn_s_wait_asynccnt)
  __builtin_amdgcn_s_wait_asynccnt(0);
#else
  asm volatile("s_wait_asynccnt 0" ::: "memory");
#endif
#endif
}

__device__ __forceinline__ f32x8 f32x8_zero() {
  f32x8 z;
#pragma unroll
  for (int i = 0; i < 8; ++i) z[i] = 0.0f;
  return z;
}

__device__ __forceinline__ bf16x16 cat8(bf16x8 lo, bf16x8 hi8) {
  return __builtin_shufflevector(lo, hi8, 0,1,2,3,4,5,6,7,8,9,10,11,12,13,14,15);
}

__device__ __forceinline__ f32x8 wmma_bf16(bf16x16 a, bf16x16 b, f32x8 c) {
  // (neg_a, A, neg_b, B, c_mod, C, reuse_a, reuse_b)
  return __builtin_amdgcn_wmma_f32_16x16x32_bf16(false, a, false, b, (short)0, c, false, false);
}

// ---------------------------------------------------------------------------
// Kernel 1: fp32 -> bf16 conversion (vectorized, grid-stride over 8-elem chunks)
// ---------------------------------------------------------------------------
__global__ void k_f32_to_bf16(const float* __restrict__ src,
                              __bf16* __restrict__ dst, int n8) {
  int i = blockIdx.x * blockDim.x + threadIdx.x;
  int stride = gridDim.x * blockDim.x;
  for (; i < n8; i += stride) {
    const float4* s = (const float4*)(src + (size_t)i * 8);
    float4 a = s[0], b = s[1];
    bf16x8 o;
    o[0] = (__bf16)a.x; o[1] = (__bf16)a.y; o[2] = (__bf16)a.z; o[3] = (__bf16)a.w;
    o[4] = (__bf16)b.x; o[5] = (__bf16)b.y; o[6] = (__bf16)b.z; o[7] = (__bf16)b.w;
    *(bf16x8*)(dst + (size_t)i * 8) = o;
  }
}

// ---------------------------------------------------------------------------
// Kernel 2/4: tiled bf16 GEMM  C = A(MxK) @ B(NxK)^T, f32 accumulate
// WG tile 128x128 (8 waves as 2(M) x 4(N); each wave 64x32 = 4x2 wmma tiles).
// Double-buffered K-tiles (2 x 16 KB LDS) filled by async global->LDS copies:
// tile t+1 streams in while tile t feeds the wmma pipe; one barrier per step.
// mode 0: write f32 C[M][N]
// mode 1: QKV epilogue -> Q/K [B,H,S,HD] bf16, V transposed VT [B,H,HD,S] bf16
// ---------------------------------------------------------------------------
#define BM 128
#define BN 128
#define BK 32

__global__ __launch_bounds__(256)
void k_gemm_bf16(const __bf16* __restrict__ A, const __bf16* __restrict__ Bm,
                 int M, int N, int K, int mode,
                 float* __restrict__ Cout,
                 __bf16* __restrict__ Qo, __bf16* __restrict__ Ko,
                 __bf16* __restrict__ VTo)
{
  __shared__ __align__(16) __bf16 As[2][BM][BK];   // 16 KB
  __shared__ __align__(16) __bf16 Bs[2][BN][BK];   // 16 KB

  const int tid   = threadIdx.x;
  const int wid   = tid >> 5;
  const int lane  = tid & 31;
  const int l     = lane & 15;
  const int hi    = lane >> 4;
  const int waveM = wid & 1;        // 0..1 -> 64 rows each
  const int waveN = wid >> 1;       // 0..3 -> 32 cols each

  const int m_blk = blockIdx.y * BM;
  const int n_blk = blockIdx.x * BN;
  const int NT    = K / BK;

  auto issue_tile = [&](int t, int buf) {
#pragma unroll
    for (int i = 0; i < 2; ++i) {
      int c = tid + i * 256;
      int r = c >> 2, kk = (c & 3) * 8;
      cp16_g2l(&As[buf][r][kk], &A[(size_t)(m_blk + r) * K + t * BK + kk]);
      cp16_g2l(&Bs[buf][r][kk], &Bm[(size_t)(n_blk + r) * K + t * BK + kk]);
    }
  };

  f32x8 acc[4][2];
#pragma unroll
  for (int i = 0; i < 4; ++i)
#pragma unroll
    for (int j = 0; j < 2; ++j) acc[i][j] = f32x8_zero();

  issue_tile(0, 0);

  for (int t = 0; t < NT; ++t) {
    const int cur = t & 1;
    cp_wait_all();        // own async copies for tile t have landed
    __syncthreads();      // whole WG done writing tile t / reading buf cur^1
    if (t + 1 < NT) issue_tile(t + 1, cur ^ 1);   // overlap with compute below
    if (t + 2 < NT) {     // prefetch tile t+2 -> global_prefetch_b8
      int r = tid >> 2, kk = (tid & 3) * 8;
      __builtin_prefetch(&A[(size_t)(m_blk + r) * K + (t + 2) * BK + kk], 0, 3);
      __builtin_prefetch(&Bm[(size_t)(n_blk + r) * K + (t + 2) * BK + kk], 0, 3);
    }

    // A-fragment: lane(hi,l) row l of its 16-row tile;
    //   elems 0..7 -> K = hi*8+0..7, elems 8..15 -> K = 16+hi*8+0..7
    bf16x16 aF[4];
#pragma unroll
    for (int tt = 0; tt < 4; ++tt) {
      int row = waveM * 64 + tt * 16 + l;
      aF[tt] = cat8(*(const bf16x8*)&As[cur][row][hi * 8],
                    *(const bf16x8*)&As[cur][row][16 + hi * 8]);
    }
    // B-fragment: lane(hi,l) col l; elems 0..7 -> K = hi*8+0..7, elems 8..15 -> K = 16+hi*8+0..7
    bf16x16 bF[2];
#pragma unroll
    for (int tt = 0; tt < 2; ++tt) {
      int row = waveN * 32 + tt * 16 + l;
      bF[tt] = cat8(*(const bf16x8*)&Bs[cur][row][hi * 8],
                    *(const bf16x8*)&Bs[cur][row][16 + hi * 8]);
    }
#pragma unroll
    for (int tm = 0; tm < 4; ++tm)
#pragma unroll
      for (int tn = 0; tn < 2; ++tn)
        acc[tm][tn] = wmma_bf16(aF[tm], bF[tn], acc[tm][tn]);
  }

  // epilogue; D elem r -> row M = r + 8*hi, col N = l
#pragma unroll
  for (int tm = 0; tm < 4; ++tm)
#pragma unroll
    for (int tn = 0; tn < 2; ++tn)
#pragma unroll
      for (int r = 0; r < 8; ++r) {
        int m = m_blk + waveM * 64 + tm * 16 + r + 8 * hi;
        int n = n_blk + waveN * 32 + tn * 16 + l;
        float v = acc[tm][tn][r];
        if (mode == 0) {
          Cout[(size_t)m * N + n] = v;
        } else {
          int which = n >> 10;          // 0=q 1=k 2=v
          int d  = n & 1023;
          int h  = d >> 6;
          int e  = d & 63;
          int b  = m >> 11;
          int s  = m & 2047;
          int bh = b * NH + h;
          __bf16 bv = (__bf16)v;
          if (which == 0)      Qo[((size_t)bh * SEQ + s) * HD + e] = bv;
          else if (which == 1) Ko[((size_t)bh * SEQ + s) * HD + e] = bv;
          else                 VTo[((size_t)bh * HD + e) * SEQ + s] = bv;
        }
      }
}

// ---------------------------------------------------------------------------
// Kernel 3: flash attention, double-buffered LDS K/V tiles (64 kv cols/step).
// 8 waves x 16 query rows = 128 q rows per WG. Online softmax, all wmma bf16.
// Async copies stream tile t+1 while tile t is computed; 1 barrier per step.
// grid: (SEQ/128, B*H), block 256.
// ---------------------------------------------------------------------------
__global__ __launch_bounds__(256)
void k_attention(const __bf16* __restrict__ Q, const __bf16* __restrict__ Kc,
                 const __bf16* __restrict__ VT, __bf16* __restrict__ Attn)
{
  __shared__ __align__(16) __bf16 Ks[2][64][HD];     // [buf][kv][e]  16 KB
  __shared__ __align__(16) __bf16 Vs[2][HD][64];     // [buf][e][kv]  16 KB
  __shared__ __align__(16) __bf16 Pbuf[8][16][64];   // per-wave P,   16 KB

  const int tid  = threadIdx.x;
  const int wid  = tid >> 5;
  const int lane = tid & 31;
  const int l    = lane & 15;
  const int hi   = lane >> 4;

  const int bh = blockIdx.y;
  const int b  = bh >> 4;
  const int h  = bh & 15;
  const int q0 = blockIdx.x * 128 + wid * 16;

  const __bf16* qp = Q  + (size_t)bh * SEQ * HD;
  const __bf16* kp = Kc + (size_t)bh * SEQ * HD;
  const __bf16* vp = VT + (size_t)bh * HD * SEQ;

  auto issue_kv = [&](int kv0, int buf) {
#pragma unroll
    for (int i = 0; i < 2; ++i) {
      int c = tid + i * 256;
      int r = c >> 3, kk = (c & 7) * 8;
      cp16_g2l(&Ks[buf][r][kk], &kp[(size_t)(kv0 + r) * HD + kk]);
      cp16_g2l(&Vs[buf][r][kk], &vp[(size_t)r * SEQ + kv0 + kk]);
    }
  };

  // Q A-fragments (16 rows x K=64 -> two K=32 chunks), kept in registers
  bf16x16 aQ[2];
  {
    const __bf16* qr = qp + (size_t)(q0 + l) * HD;
#pragma unroll
    for (int c = 0; c < 2; ++c)
      aQ[c] = cat8(*(const bf16x8*)&qr[c * 32 + hi * 8],
                   *(const bf16x8*)&qr[c * 32 + 16 + hi * 8]);
  }

  float m_run[8], l_run[8];
#pragma unroll
  for (int r = 0; r < 8; ++r) { m_run[r] = -3.0e38f; l_run[r] = 0.0f; }
  f32x8 accO[4];
#pragma unroll
  for (int nn = 0; nn < 4; ++nn) accO[nn] = f32x8_zero();

  __bf16 (*Pb)[64] = Pbuf[wid];

  const int NT = SEQ / 64;
  issue_kv(0, 0);

  for (int t = 0; t < NT; ++t) {
    const int cur = t & 1;
    cp_wait_all();
    __syncthreads();
    if (t + 1 < NT) issue_kv((t + 1) * 64, cur ^ 1);   // overlap with compute

    // ---- scores: 16 x 64, four N-subtiles, K=64 via two wmma each ----
    f32x8 sc[4];
#pragma unroll
    for (int n = 0; n < 4; ++n) {
      int row = n * 16 + l;
      f32x8 z = f32x8_zero();
#pragma unroll
      for (int c = 0; c < 2; ++c)
        z = wmma_bf16(aQ[c],
                      cat8(*(const bf16x8*)&Ks[cur][row][c * 32 + hi * 8],
                           *(const bf16x8*)&Ks[cur][row][c * 32 + 16 + hi * 8]),
                      z);
      sc[n] = z;
    }

    // ---- online softmax (row M = r + 8*hi lives across lanes of one half) ----
#pragma unroll
    for (int r = 0; r < 8; ++r) {
      float s0 = sc[0][r] * 0.125f, s1 = sc[1][r] * 0.125f;
      float s2 = sc[2][r] * 0.125f, s3 = sc[3][r] * 0.125f;
      float v = fmaxf(fmaxf(s0, s1), fmaxf(s2, s3));
#pragma unroll
      for (int off = 1; off < 16; off <<= 1)
        v = fmaxf(v, __shfl_xor(v, off, 32));
      float mnew = fmaxf(m_run[r], v);
      float resc = __expf(m_run[r] - mnew);
      m_run[r] = mnew;
      float p0 = __expf(s0 - mnew), p1 = __expf(s1 - mnew);
      float p2 = __expf(s2 - mnew), p3 = __expf(s3 - mnew);
      l_run[r] = l_run[r] * resc + (p0 + p1) + (p2 + p3);
#pragma unroll
      for (int nn = 0; nn < 4; ++nn) accO[nn][r] *= resc;
      int mrow = r + 8 * hi;
      Pb[mrow][l]      = (__bf16)p0;
      Pb[mrow][16 + l] = (__bf16)p1;
      Pb[mrow][32 + l] = (__bf16)p2;
      Pb[mrow][48 + l] = (__bf16)p3;
    }

    // all 32 lanes' ds_stores must land before re-reading as A-fragment
    asm volatile("s_wait_dscnt 0" ::: "memory");

    bf16x16 pA[2];
#pragma unroll
    for (int c = 0; c < 2; ++c)
      pA[c] = cat8(*(const bf16x8*)&Pb[l][c * 32 + hi * 8],
                   *(const bf16x8*)&Pb[l][c * 32 + 16 + hi * 8]);

    // ---- PV: out += P(16x64) @ V(64x64); V cols contiguous thanks to VT ----
#pragma unroll
    for (int nn = 0; nn < 4; ++nn) {
      int row = nn * 16 + l;
#pragma unroll
      for (int c = 0; c < 2; ++c)
        accO[nn] = wmma_bf16(pA[c],
                             cat8(*(const bf16x8*)&Vs[cur][row][c * 32 + hi * 8],
                                  *(const bf16x8*)&Vs[cur][row][c * 32 + 16 + hi * 8]),
                             accO[nn]);
    }
  }

  // ---- finalize: reduce row sums across the half, normalize, store bf16 ----
  float inv[8];
#pragma unroll
  for (int r = 0; r < 8; ++r) {
    float s = l_run[r];
#pragma unroll
    for (int off = 1; off < 16; off <<= 1) s += __shfl_xor(s, off, 32);
    inv[r] = 1.0f / s;
  }
#pragma unroll
  for (int nn = 0; nn < 4; ++nn)
#pragma unroll
    for (int r = 0; r < 8; ++r) {
      int srow = q0 + r + 8 * hi;
      int feat = h * HD + nn * 16 + l;
      Attn[((size_t)b * SEQ + srow) * EMB + feat] = (__bf16)(accO[nn][r] * inv[r]);
    }
}

// ---------------------------------------------------------------------------
// launcher
// ---------------------------------------------------------------------------
extern "C" void kernel_launch(void* const* d_in, const int* in_sizes, int n_in,
                              void* d_out, int out_size, void* d_ws, size_t ws_size,
                              hipStream_t stream) {
  const float* x     = (const float*)d_in[0];
  const float* w_qkv = (const float*)d_in[1];
  const float* w_out = (const float*)d_in[2];
  float* out = (float*)d_out;

  char* ws = (char*)d_ws;
  size_t off = 0;
  auto take = [&](size_t elems) {
    __bf16* p = (__bf16*)(ws + off);
    off += elems * sizeof(__bf16);
    return p;
  };
  const size_t nX  = (size_t)B_SZ * SEQ * EMB;   // 4M
  const size_t nWq = (size_t)3 * EMB * EMB;      // 3M
  const size_t nWo = (size_t)EMB * EMB;          // 1M
  const size_t nQ  = (size_t)B_SZ * NH * SEQ * HD;

  __bf16* xb   = take(nX);
  __bf16* wqb  = take(nWq);
  __bf16* wob  = take(nWo);
  __bf16* Qb   = take(nQ);
  __bf16* Kb   = take(nQ);
  __bf16* VTb  = take(nQ);
  __bf16* Attn = take(nX);

  // 1) fp32 -> bf16
  k_f32_to_bf16<<<1024, 256, 0, stream>>>(x,     xb,  (int)(nX  / 8));
  k_f32_to_bf16<<<1024, 256, 0, stream>>>(w_qkv, wqb, (int)(nWq / 8));
  k_f32_to_bf16<<<512,  256, 0, stream>>>(w_out, wob, (int)(nWo / 8));

  // 2) QKV GEMM: [4096,1024] @ [3072,1024]^T, epilogue scatters Q/K/VT
  {
    dim3 grid((3 * EMB) / BN, (B_SZ * SEQ) / BM);   // (24, 32)
    k_gemm_bf16<<<grid, 256, 0, stream>>>(xb, wqb, B_SZ * SEQ, 3 * EMB, EMB,
                                          /*mode=*/1, nullptr, Qb, Kb, VTb);
  }

  // 3) flash attention
  {
    dim3 grid(SEQ / 128, B_SZ * NH);                // (16, 32)
    k_attention<<<grid, 256, 0, stream>>>(Qb, Kb, VTb, Attn);
  }

  // 4) out projection: [4096,1024] @ [1024,1024]^T -> f32 d_out
  {
    dim3 grid(EMB / BN, (B_SZ * SEQ) / BM);         // (8, 32)
    k_gemm_bf16<<<grid, 256, 0, stream>>>(Attn, wob, B_SZ * SEQ, EMB, EMB,
                                          /*mode=*/0, out, nullptr, nullptr, nullptr);
  }
}
